// HyperDiffuseNet_83880711291225
// MI455X (gfx1250) — matrix operand order repaired and checked by reading, not compiled, not run on hardware
//
#include <hip/hip_runtime.h>
#include <hip/hip_bf16.h>

// ---------------------------------------------------------------------------
// HyperDiffuseNet on MI455X (gfx1250, wave32, WMMA).
// Dominant GEMMs run through v_wmma_f32_16x16x32_bf16 (fp32 accumulate).
// ---------------------------------------------------------------------------

typedef __attribute__((ext_vector_type(16))) __bf16 v16bf;
typedef __attribute__((ext_vector_type(8)))  __bf16 v8bf;
typedef __attribute__((ext_vector_type(8)))  float  v8f;

constexpr int NN   = 6144;   // rows / attn dim
constexpr int JJ   = 2000;   // gene dim
constexpr int HH   = 100;    // hidden
constexpr int OO   = 6144;   // value dim
constexpr int KXP  = 2016;   // J padded to mult of 32
constexpr int KQK  = 128;    // H padded to mult of 32 (x4 wmma K-steps)
constexpr int KLOG = KXP + KQK; // 2144
constexpr int LDK  = 40;     // LDS K-stride (pad 32 -> 40 bf16, breaks bank conflicts)

// ------------------------------- small kernels ------------------------------

__global__ __launch_bounds__(256) void row_stats_kernel(
    const float* __restrict__ X, const float* __restrict__ spatial,
    float* __restrict__ inv_norm, float* __restrict__ sqv,
    float* __restrict__ sxv, float* __restrict__ syv) {
  int i = blockIdx.x, t = threadIdx.x;
  const float4* xr = (const float4*)(X + (size_t)i * JJ);   // 2000 = 500 float4
  float s = 0.f;
  for (int c = t; c < JJ / 4; c += 256) {
    float4 v = xr[c];
    s += v.x * v.x + v.y * v.y + v.z * v.z + v.w * v.w;
  }
  __shared__ float red[256];
  red[t] = s; __syncthreads();
  for (int off = 128; off; off >>= 1) {
    if (t < off) red[t] += red[t + off];
    __syncthreads();
  }
  if (t == 0) {
    float n = sqrtf(red[0]);
    inv_norm[i] = 1.f / fmaxf(n, 1e-12f);
    float a = spatial[i * 2 + 0], b = spatial[i * 2 + 1];
    sxv[i] = a; syv[i] = b; sqv[i] = a * a + b * b;
  }
}

// one wave per row; H=100 parallel over lanes (h = lane, lane+32, ...)
__global__ __launch_bounds__(256) void mlp_f_kernel(
    const float* __restrict__ spatial, const float* __restrict__ Ws,
    const float* __restrict__ bsv, const float* __restrict__ W1,
    const float* __restrict__ b1, const float* __restrict__ W2,
    const float* __restrict__ b2, const float* __restrict__ Wf,
    const float* __restrict__ bfv, float* __restrict__ fout) {
  __shared__ float sp[8][104];
  __shared__ float tb[8][104];
  int w = threadIdx.x >> 5, lane = threadIdx.x & 31;
  int row = blockIdx.x * 8 + w;
  float x = spatial[row * 2 + 0], y = spatial[row * 2 + 1];
  for (int h = lane; h < HH; h += 32)
    sp[w][h] = x * Ws[h] + y * Ws[HH + h] + bsv[h];
  __syncthreads();
  for (int h = lane; h < HH; h += 32) {
    float acc = b1[h];
    for (int k = 0; k < HH; ++k) acc += sp[w][k] * W1[k * HH + h];
    tb[w][h] = tanhf(acc);
  }
  __syncthreads();
  float partial = 0.f;
  for (int h = lane; h < HH; h += 32) {
    float acc = b2[h];
    for (int k = 0; k < HH; ++k) acc += tb[w][k] * W2[k * HH + h];
    partial += fmaxf(acc, 0.f) * Wf[h];
  }
  for (int off = 16; off; off >>= 1) partial += __shfl_down(partial, off, 32);
  if (lane == 0) fout[row] = partial + bfv[0];
}

__global__ __launch_bounds__(256) void convX_kernel(
    const float* __restrict__ X, const float* __restrict__ inv_norm,
    __bf16* __restrict__ Xb, __bf16* __restrict__ Xn) {
  size_t idx = (size_t)blockIdx.x * 256 + threadIdx.x;  // N*KXP total
  int i = (int)(idx / KXP), k = (int)(idx % KXP);
  float v = (k < JJ) ? X[(size_t)i * JJ + k] : 0.f;
  Xb[idx] = (__bf16)v;
  Xn[idx] = (__bf16)(v * inv_norm[i]);
}

__global__ __launch_bounds__(256) void convWv_kernel(
    const float* __restrict__ Wv, __bf16* __restrict__ Wvb) {
  size_t idx = (size_t)blockIdx.x * 256 + threadIdx.x;  // KXP*OO total
  int k = (int)(idx / OO), o = (int)(idx % OO);
  float v = (k < JJ) ? Wv[(size_t)k * OO + o] : 0.f;
  Wvb[idx] = (__bf16)v;
}

// Q/sqrt(dk) and K projections (fp32 reduce, bf16 store, H padded to 128)
__global__ __launch_bounds__(128) void qk_kernel(
    const float* __restrict__ X, const float* __restrict__ Wq,
    const float* __restrict__ bq, const float* __restrict__ Wk,
    const float* __restrict__ bk, __bf16* __restrict__ Qs,
    __bf16* __restrict__ Kb) {
  int i = blockIdx.x, h = threadIdx.x;  // 128 threads
  bool valid = h < HH;
  int hh = valid ? h : 0;               // OOB-safe column
  __shared__ float xs[128];
  float q = 0.f, k = 0.f;
  for (int j0 = 0; j0 < JJ; j0 += 128) {
    int take = (JJ - j0 < 128) ? (JJ - j0) : 128;
    __syncthreads();
    if (h < take) xs[h] = X[(size_t)i * JJ + j0 + h];
    __syncthreads();
    for (int jj = 0; jj < take; ++jj) {
      float x = xs[jj];
      int j = j0 + jj;
      q += x * Wq[(size_t)j * HH + hh];
      k += x * Wk[(size_t)j * HH + hh];
    }
  }
  float qv = valid ? (q + bq[hh]) * 0.1f : 0.f;  // /sqrt(100)
  float kv = valid ? (k + bk[hh]) : 0.f;
  Qs[(size_t)i * KQK + h] = (__bf16)qv;
  Kb[(size_t)i * KQK + h] = (__bf16)kv;
}

// -------------------------------- WMMA GEMM ---------------------------------
// 128x128 block, BK=32, 8 waves (2x4), each wave 64x32 = 4x2 wmma frags.
enum GemmMode { GM_LOGITS = 0, GM_V = 1, GM_OUT = 2 };

template <int MODE>
__global__ __launch_bounds__(256) void wmma_gemm_kernel(
    const __bf16* __restrict__ Xn, const __bf16* __restrict__ Qs,
    const __bf16* __restrict__ Kb,              // logits mode operands
    const __bf16* __restrict__ Abf,             // V mode: Xb [m][k]
    const __bf16* __restrict__ Bsrc,            // V: Wvb [k][n]; OUT: Vb [k][n]
    const float* __restrict__ Afp,              // OUT mode: attn fp32 [m][k]
    const float* __restrict__ bv,               // V mode bias
    __bf16* __restrict__ outB,                  // V mode output (bf16)
    float* __restrict__ outF,                   // logits / out output (fp32)
    int Kdim) {
  __shared__ __align__(16) __bf16 As[128 * LDK];
  __shared__ __align__(16) __bf16 Bs[128 * LDK];
  const int tid  = threadIdx.x;
  const int lane = tid & 31;
  const int wv   = tid >> 5;
  const int wm   = (wv & 1) * 64;
  const int wn   = (wv >> 1) * 32;
  const int nb   = (blockIdx.x % (NN / 128)) * 128;
  const int mb   = (blockIdx.x / (NN / 128)) * 128;

  v8f c[4][2];
#pragma unroll
  for (int q = 0; q < 4; ++q)
#pragma unroll
    for (int p = 0; p < 2; ++p)
#pragma unroll
      for (int e = 0; e < 8; ++e) c[q][p][e] = 0.f;

  for (int k0 = 0; k0 < Kdim; k0 += 32) {
    // ---- stage A tile (128 x 32 bf16, [m][k]) ----
#pragma unroll
    for (int it = 0; it < 2; ++it) {
      int ch  = tid + 256 * it;     // 512 chunks of 8 bf16
      int row = ch >> 2;
      int kc  = (ch & 3) * 8;
      v8bf av;
      if (MODE == GM_LOGITS) {
        if (k0 < KXP)
          av = *(const v8bf*)(Xn + (size_t)(mb + row) * KXP + k0 + kc);
        else
          av = *(const v8bf*)(Qs + (size_t)(mb + row) * KQK + (k0 - KXP) + kc);
      } else if (MODE == GM_V) {
        av = *(const v8bf*)(Abf + (size_t)(mb + row) * KXP + k0 + kc);
      } else {  // GM_OUT: fp32 attn -> bf16 on the fly
        const float* ap = Afp + (size_t)(mb + row) * NN + k0 + kc;
#pragma unroll
        for (int e = 0; e < 8; ++e) av[e] = (__bf16)ap[e];
      }
      *(v8bf*)(&As[row * LDK + kc]) = av;
    }
    // ---- stage B tile into [n][k] layout ----
    if (MODE == GM_LOGITS) {
#pragma unroll
      for (int it = 0; it < 2; ++it) {
        int ch  = tid + 256 * it;
        int row = ch >> 2;
        int kc  = (ch & 3) * 8;
        v8bf bb;
        if (k0 < KXP)
          bb = *(const v8bf*)(Xn + (size_t)(nb + row) * KXP + k0 + kc);
        else
          bb = *(const v8bf*)(Kb + (size_t)(nb + row) * KQK + (k0 - KXP) + kc);
        *(v8bf*)(&Bs[row * LDK + kc]) = bb;
      }
    } else {  // transpose-stage from row-major [k][n]
#pragma unroll
      for (int it = 0; it < 2; ++it) {
        int ch = tid + 256 * it;
        int kk = ch >> 4;          // 0..31
        int nc = (ch & 15) * 8;    // 0..120
        v8bf bb = *(const v8bf*)(Bsrc + (size_t)(k0 + kk) * NN + nb + nc);
        if (k0 + 32 < Kdim)        // prefetch next K tile -> global_prefetch_b8
          __builtin_prefetch(Bsrc + (size_t)(k0 + 32 + kk) * NN + nb + nc, 0, 0);
#pragma unroll
        for (int e = 0; e < 8; ++e) Bs[(nc + e) * LDK + kk] = bb[e];
      }
    }
    __syncthreads();

    // ---- fragments per ISA 16-bit layouts ----
    const int ml = lane & 15;
    const int ka = (lane < 16) ? 0 : 8;   // A: K {0..7,16..23} / {8..15,24..31}
    const int kb = (lane < 16) ? 0 : 8;   // B: K {0..7,16..23} / {8..15,24..31}
    v16bf a[4], b[2];
#pragma unroll
    for (int q = 0; q < 4; ++q) {
      const __bf16* p = &As[(wm + 16 * q + ml) * LDK + ka];
      v8bf lo = *(const v8bf*)p;
      v8bf hi = *(const v8bf*)(p + 16);
      a[q] = __builtin_shufflevector(lo, hi, 0, 1, 2, 3, 4, 5, 6, 7, 8, 9, 10,
                                     11, 12, 13, 14, 15);
    }
#pragma unroll
    for (int p2 = 0; p2 < 2; ++p2) {
      const __bf16* p = &Bs[(wn + 16 * p2 + ml) * LDK + kb];
      v8bf lo = *(const v8bf*)p;
      v8bf hi = *(const v8bf*)(p + 16);
      b[p2] = __builtin_shufflevector(lo, hi, 0, 1, 2, 3, 4, 5, 6, 7, 8, 9, 10,
                                      11, 12, 13, 14, 15);
    }
#pragma unroll
    for (int q = 0; q < 4; ++q)
#pragma unroll
      for (int p2 = 0; p2 < 2; ++p2)
        c[q][p2] = __builtin_amdgcn_wmma_f32_16x16x32_bf16(
            false, a[q], false, b[p2], (short)0, c[q][p2], false, false);

    __syncthreads();
  }

  // ---- epilogue: C frag layout (VGPR r, lane l): M = r + 8*(l>=16), N = l&15
  const int half = lane >> 4;
  const int nl   = lane & 15;
#pragma unroll
  for (int p2 = 0; p2 < 2; ++p2) {
    int gn = nb + wn + 16 * p2 + nl;
    float bias = (MODE == GM_V) ? bv[gn] : 0.f;
#pragma unroll
    for (int q = 0; q < 4; ++q) {
#pragma unroll
      for (int r = 0; r < 8; ++r) {
        int gm = mb + wm + 16 * q + r + 8 * half;
        float val = c[q][p2][r] + bias;
        if (MODE == GM_V)
          outB[(size_t)gm * NN + gn] = (__bf16)val;
        else
          outF[(size_t)gm * NN + gn] = val;
      }
    }
  }
}

// ------------------------------- softmax ------------------------------------
// Fuses gaussian spatial_sim + tanh fdiff into the logits, stable softmax,
// writes attn in place (row staged in 24KB LDS).
__global__ __launch_bounds__(256) void softmax_kernel(
    float* __restrict__ attn, const float* __restrict__ sqv,
    const float* __restrict__ sxv, const float* __restrict__ syv,
    const float* __restrict__ fv) {
  int i = blockIdx.x, t = threadIdx.x;
  __shared__ float buf[NN];
  __shared__ float red[256];
  float sqi = sqv[i], sxi = sxv[i], syi = syv[i], fi = fv[i];
  float* row = attn + (size_t)i * NN;
  float m = -3.4e38f;
  for (int j = t; j < NN; j += 256) {
    float g  = row[j];
    float d2 = fmaxf(sqi + sqv[j] - 2.f * (sxi * sxv[j] + syi * syv[j]), 0.f);
    float lv = g + __expf(-0.5f * d2) + tanhf(fi - fv[j]);
    buf[j] = lv;
    m = fmaxf(m, lv);
  }
  red[t] = m; __syncthreads();
  for (int off = 128; off; off >>= 1) {
    if (t < off) red[t] = fmaxf(red[t], red[t + off]);
    __syncthreads();
  }
  m = red[0]; __syncthreads();
  float s = 0.f;
  for (int j = t; j < NN; j += 256) {
    float e = __expf(buf[j] - m);
    buf[j] = e;
    s += e;
  }
  red[t] = s; __syncthreads();
  for (int off = 128; off; off >>= 1) {
    if (t < off) red[t] += red[t + off];
    __syncthreads();
  }
  float inv = 1.f / red[0];
  for (int j = t; j < NN; j += 256) row[j] = buf[j] * inv;
}

// ------------------------------- launcher -----------------------------------

extern "C" void kernel_launch(void* const* d_in, const int* in_sizes, int n_in,
                              void* d_out, int out_size, void* d_ws,
                              size_t ws_size, hipStream_t stream) {
  const float* gene    = (const float*)d_in[0];
  const float* spatial = (const float*)d_in[1];
  const float* Wq = (const float*)d_in[2];
  const float* bq = (const float*)d_in[3];
  const float* Wk = (const float*)d_in[4];
  const float* bk = (const float*)d_in[5];
  const float* Wv = (const float*)d_in[6];
  const float* bv = (const float*)d_in[7];
  const float* Ws = (const float*)d_in[8];
  const float* bs = (const float*)d_in[9];
  const float* W1 = (const float*)d_in[10];
  const float* b1 = (const float*)d_in[11];
  const float* W2 = (const float*)d_in[12];
  const float* b2 = (const float*)d_in[13];
  const float* Wf = (const float*)d_in[14];
  const float* bf = (const float*)d_in[15];

  // workspace partition (~153 MB total)
  char*  ws  = (char*)d_ws;
  size_t off = 0;
  auto take = [&](size_t bytes) {
    char* p = ws + off;
    off += (bytes + 255) & ~(size_t)255;
    return p;
  };
  __bf16* Xb  = (__bf16*)take((size_t)NN * KXP * 2);   // raw X, bf16, padded
  __bf16* Xn  = (__bf16*)take((size_t)NN * KXP * 2);   // row-normalized X
  __bf16* Wvb = (__bf16*)take((size_t)KXP * OO * 2);   // Wv bf16, padded
  __bf16* Vb  = (__bf16*)take((size_t)NN * OO * 2);    // V = X@Wv + bv, bf16
  __bf16* Qs  = (__bf16*)take((size_t)NN * KQK * 2);   // (Q+bq)/sqrt(dk)
  __bf16* Kb  = (__bf16*)take((size_t)NN * KQK * 2);   // K+bk
  float* inv_norm = (float*)take((size_t)NN * 4);
  float* sqv = (float*)take((size_t)NN * 4);
  float* sxv = (float*)take((size_t)NN * 4);
  float* syv = (float*)take((size_t)NN * 4);
  float* fv  = (float*)take((size_t)NN * 4);

  float* attn = (float*)d_out;           // [N,N]
  float* outO = attn + (size_t)NN * NN;  // [N,O]

  const int GEMM_BLOCKS = (NN / 128) * (NN / 128);  // 2304

  row_stats_kernel<<<NN, 256, 0, stream>>>(gene, spatial, inv_norm, sqv, sxv, syv);
  mlp_f_kernel<<<NN / 8, 256, 0, stream>>>(spatial, Ws, bs, W1, b1, W2, b2, Wf, bf, fv);
  convX_kernel<<<(NN * KXP) / 256, 256, 0, stream>>>(gene, inv_norm, Xb, Xn);
  convWv_kernel<<<(KXP * OO) / 256, 256, 0, stream>>>(Wv, Wvb);
  qk_kernel<<<NN, 128, 0, stream>>>(gene, Wq, bq, Wk, bk, Qs, Kb);

  // V = Xb @ Wvb + bv  (bf16 out)
  wmma_gemm_kernel<GM_V><<<GEMM_BLOCKS, 256, 0, stream>>>(
      nullptr, nullptr, nullptr, Xb, Wvb, nullptr, bv, Vb, nullptr, KXP);
  // logits(matmul part) = [Xn|Q/dk] @ [Xn|K]^T  -> d_out attn region (fp32)
  wmma_gemm_kernel<GM_LOGITS><<<GEMM_BLOCKS, 256, 0, stream>>>(
      Xn, Qs, Kb, nullptr, nullptr, nullptr, nullptr, nullptr, attn, KLOG);
  // fused spatial_sim + fdiff + stable softmax, in place
  softmax_kernel<<<NN, 256, 0, stream>>>(attn, sqv, sxv, syv, fv);
  // out = attn @ V   (attn fp32 -> bf16 during LDS staging)
  wmma_gemm_kernel<GM_OUT><<<GEMM_BLOCKS, 256, 0, stream>>>(
      nullptr, nullptr, nullptr, nullptr, Vb, attn, nullptr, nullptr, outO, NN);
}
